// TPILlamaAttention_1005022347443
// MI455X (gfx1250) — matrix operand order repaired and checked by reading, not compiled, not run on hardware
//
#include <hip/hip_runtime.h>

#define SLEN 2048
#define BATCH 2
#define HIDN 4096
#define NH 32
#define NKV 8
#define HD 128

typedef __bf16 bf16_t;
typedef __attribute__((ext_vector_type(16))) __bf16 v16bf;
typedef __attribute__((ext_vector_type(8)))  float  v8f;

#define WMMA_BF16(a, b, c) \
  __builtin_amdgcn_wmma_f32_16x16x32_bf16(false, (a), false, (b), (short)0, (c), false, false)

static __device__ __forceinline__ v8f zero_v8f() {
  v8f z;
#pragma unroll
  for (int i = 0; i < 8; ++i) z[i] = 0.0f;
  return z;
}

// A-fragment (16x32 bf16, row-major source, row stride ld elements):
// lane L: row = L%16; VGPR0..3 hold K = k0..k0+7, VGPR4..7 hold K = k0+16..k0+23,
// with k0 = (L/16)*8  -> two 16B contiguous loads.
static __device__ __forceinline__ v16bf load_frag_a(const bf16_t* p, int ld, int lane) {
  const bf16_t* q = p + (size_t)(lane & 15) * ld + ((lane >> 4) << 3);
  union { v16bf v; uint4 u[2]; } U;
  U.u[0] = *(const uint4*)q;
  U.u[1] = *(const uint4*)(q + 16);
  return U.v;
}

// B-fragment (32x16 bf16). Source is the [N x K] row-major matrix (lane = column N
// of B = row N of source). Lane L holds K = k0..k0+7 and k0+16..k0+23, k0 = (L/16)*8.
static __device__ __forceinline__ v16bf load_frag_b(const bf16_t* p, int ld, int lane) {
  const bf16_t* q = p + (size_t)(lane & 15) * ld + ((lane >> 4) << 3);
  union { v16bf v; uint4 u[2]; } U;
  U.u[0] = *(const uint4*)q;
  U.u[1] = *(const uint4*)(q + 16);
  return U.v;
}

// ---------------------------------------------------------------------------
// fp32 -> bf16 convert (vectorized by 4)
// ---------------------------------------------------------------------------
__global__ void cvt_f32_to_bf16(const float* __restrict__ in, bf16_t* __restrict__ out, int n4) {
  int i = blockIdx.x * blockDim.x + threadIdx.x;
  if (i >= n4) return;
  float4 v = ((const float4*)in)[i];
  union { bf16_t h[4]; uint2 u; } o;
  o.h[0] = (bf16_t)v.x; o.h[1] = (bf16_t)v.y;
  o.h[2] = (bf16_t)v.z; o.h[3] = (bf16_t)v.w;
  ((uint2*)out)[i] = o.u;
}

// ---------------------------------------------------------------------------
// WMMA GEMM: C[M,N] = A[M,K] * W[N,K]^T  (A, W bf16 row-major along K)
// MODE 0: store fp32 row-major (ldc). MODE 2: store bf16 transposed V layout
//         dst[(b*NKV*HD + n) * SLEN + s]  with m = b*SLEN + s.
// Block: 256 threads = 8 waves arranged 2(M) x 4(N); wave tile 32x64
// (2 A-frags x 4 B-frags -> 8 WMMAs per 12 b128 loads).
// ---------------------------------------------------------------------------
template <int MODE>
__global__ __launch_bounds__(256)
void gemm_bf16_kernel(const bf16_t* __restrict__ A, const bf16_t* __restrict__ W,
                      void* __restrict__ Cout, int K, int lda, int ldw, int ldc) {
  const int lane = threadIdx.x & 31;
  const int wave = threadIdx.x >> 5;
  const int m0 = blockIdx.y * 64 + (wave >> 2) * 32;
  const int n0 = blockIdx.x * 256 + (wave & 3) * 64;

  v8f c0[4], c1[4];
#pragma unroll
  for (int j = 0; j < 4; ++j) { c0[j] = zero_v8f(); c1[j] = zero_v8f(); }

  const bf16_t* A0 = A + (size_t)m0 * lda;
  const bf16_t* A1 = A + (size_t)(m0 + 16) * lda;
  const bf16_t* Wr[4];
#pragma unroll
  for (int j = 0; j < 4; ++j) Wr[j] = W + (size_t)(n0 + 16 * j) * ldw;

#pragma unroll 2
  for (int k0 = 0; k0 < K; k0 += 32) {
    // prefetch the streaming A operand one K-step ahead
    __builtin_prefetch(A0 + k0 + 32, 0, 0);
    __builtin_prefetch(A1 + k0 + 32, 0, 0);
    v16bf a0 = load_frag_a(A0 + k0, lda, lane);
    v16bf a1 = load_frag_a(A1 + k0, lda, lane);
#pragma unroll
    for (int j = 0; j < 4; ++j) {
      v16bf bj = load_frag_b(Wr[j] + k0, ldw, lane);
      c0[j] = WMMA_BF16(a0, bj, c0[j]);
      c1[j] = WMMA_BF16(a1, bj, c1[j]);
    }
  }

  const int rofs = (lane >> 4) << 3;
  const int col = lane & 15;
#pragma unroll
  for (int r = 0; r < 8; ++r) {
    int mA = m0 + rofs + r;
    int mB = mA + 16;
    if (MODE == 0) {
      float* C = (float*)Cout;
#pragma unroll
      for (int j = 0; j < 4; ++j) {
        C[(size_t)mA * ldc + n0 + 16 * j + col] = c0[j][r];
        C[(size_t)mB * ldc + n0 + 16 * j + col] = c1[j][r];
      }
    } else {  // transposed V store
      bf16_t* C = (bf16_t*)Cout;
      int bA = mA >> 11, sA = mA & 2047;
      int bB = mB >> 11, sB = mB & 2047;
#pragma unroll
      for (int j = 0; j < 4; ++j) {
        C[((size_t)(bA * (NKV * HD) + n0 + 16 * j + col)) * SLEN + sA] = (bf16_t)c0[j][r];
        C[((size_t)(bB * (NKV * HD) + n0 + 16 * j + col)) * SLEN + sB] = (bf16_t)c1[j][r];
      }
    }
  }
}

// ---------------------------------------------------------------------------
// RoPE: reads fp32 [M, nheads*HD], writes bf16 same layout. One thread per
// (token, head, d) with d in [0,64): handles the (d, d+64) rotation pair.
// ---------------------------------------------------------------------------
__global__ void rope_kernel(const float* __restrict__ X, const float* __restrict__ cosb,
                            const float* __restrict__ sinb, bf16_t* __restrict__ Y,
                            int nheads, size_t total) {
  size_t idx = blockIdx.x * (size_t)blockDim.x + threadIdx.x;
  if (idx >= total) return;
  int dp = (int)(idx & 63);
  size_t t = idx >> 6;
  int h = (int)(t % nheads);
  size_t bs = t / nheads;  // b*SLEN + s
  size_t base = bs * (size_t)(nheads * HD) + (size_t)h * HD;
  float x1 = X[base + dp];
  float x2 = X[base + dp + 64];
  size_t cb = bs * HD;
  float c1 = cosb[cb + dp],      s1 = sinb[cb + dp];
  float c2 = cosb[cb + dp + 64], s2 = sinb[cb + dp + 64];
  Y[base + dp]      = (bf16_t)(x1 * c1 - x2 * s1);
  Y[base + dp + 64] = (bf16_t)(x2 * c2 + x1 * s2);
}

// ---------------------------------------------------------------------------
// Flash attention (causal, GQA). One wave per (b, h, 16-row q tile).
// Q: bf16 [B,S,NH*HD]; K: bf16 [B,S,NKV*HD]; Vt: bf16 [B*NKV*HD, S] (transposed);
// O: bf16 [B,S,NH*HD], already divided by the softmax denominator.
// ---------------------------------------------------------------------------
__global__ __launch_bounds__(256)
void flash_attn_kernel(const bf16_t* __restrict__ Qb, const bf16_t* __restrict__ Kb,
                       const bf16_t* __restrict__ Vt, bf16_t* __restrict__ Ob) {
  __shared__ __align__(16) bf16_t psh[8][16 * 32];
  const int lane = threadIdx.x & 31;
  const int wave = threadIdx.x >> 5;
  const int gw = blockIdx.x * 8 + wave;
  const int qt = gw & 127;   // S/16 tiles
  const int bh = gw >> 7;
  const int h = bh & 31;
  const int b = bh >> 5;
  const int kvh = h >> 2;    // NH/NKV = 4
  const int q0 = qt * 16;

  const bf16_t* Qp = Qb + ((size_t)b * SLEN + q0) * (NH * HD) + h * HD;
  const bf16_t* Kp = Kb + (size_t)b * SLEN * (NKV * HD) + kvh * HD;
  const bf16_t* Vp = Vt + (size_t)(b * (NKV * HD) + kvh * HD) * SLEN;

  v16bf qf0 = load_frag_a(Qp + 0,  NH * HD, lane);
  v16bf qf1 = load_frag_a(Qp + 32, NH * HD, lane);
  v16bf qf2 = load_frag_a(Qp + 64, NH * HD, lane);
  v16bf qf3 = load_frag_a(Qp + 96, NH * HD, lane);

  v8f o[8];
#pragma unroll
  for (int d = 0; d < 8; ++d) o[d] = zero_v8f();
  float m_r[8], l_r[8];
#pragma unroll
  for (int r = 0; r < 8; ++r) { m_r[r] = -1e30f; l_r[r] = 0.0f; }

  const float scale = 0.08838834764831845f;  // 1/sqrt(128)
  const int rowbase = q0 + ((lane >> 4) << 3);
  const int c0base = lane & 15;

  for (int j = 0; j < q0 + 16; j += 32) {
    // --- S = Q * K^T for 32 keys (two 16-wide tiles) ---
    v8f s0 = zero_v8f(), s1 = zero_v8f();
    const bf16_t* K0 = Kp + (size_t)j * (NKV * HD);
    const bf16_t* K1 = K0 + (size_t)16 * (NKV * HD);
    {
      v16bf bf;
      bf = load_frag_b(K0 + 0,  NKV * HD, lane); s0 = WMMA_BF16(qf0, bf, s0);
      bf = load_frag_b(K0 + 32, NKV * HD, lane); s0 = WMMA_BF16(qf1, bf, s0);
      bf = load_frag_b(K0 + 64, NKV * HD, lane); s0 = WMMA_BF16(qf2, bf, s0);
      bf = load_frag_b(K0 + 96, NKV * HD, lane); s0 = WMMA_BF16(qf3, bf, s0);
      bf = load_frag_b(K1 + 0,  NKV * HD, lane); s1 = WMMA_BF16(qf0, bf, s1);
      bf = load_frag_b(K1 + 32, NKV * HD, lane); s1 = WMMA_BF16(qf1, bf, s1);
      bf = load_frag_b(K1 + 64, NKV * HD, lane); s1 = WMMA_BF16(qf2, bf, s1);
      bf = load_frag_b(K1 + 96, NKV * HD, lane); s1 = WMMA_BF16(qf3, bf, s1);
    }

    // --- online softmax (rows live across 16-lane halves in C layout) ---
    const int col0 = j + c0base;
    const int col1 = col0 + 16;
    float pr0[8], pr1[8];
#pragma unroll
    for (int r = 0; r < 8; ++r) {
      const int row = rowbase + r;
      float e0 = (col0 <= row) ? s0[r] * scale : -1e30f;
      float e1 = (col1 <= row) ? s1[r] * scale : -1e30f;
      float mx = fmaxf(e0, e1);
#pragma unroll
      for (int off = 1; off < 16; off <<= 1) mx = fmaxf(mx, __shfl_xor(mx, off, 32));
      const float mnew = fmaxf(m_r[r], mx);
      const float alpha = __expf(m_r[r] - mnew);
      const float p0 = __expf(e0 - mnew);
      const float p1 = __expf(e1 - mnew);
      float rs = p0 + p1;
#pragma unroll
      for (int off = 1; off < 16; off <<= 1) rs += __shfl_xor(rs, off, 32);
      l_r[r] = l_r[r] * alpha + rs;
      m_r[r] = mnew;
#pragma unroll
      for (int d = 0; d < 8; ++d) o[d][r] *= alpha;
      pr0[r] = p0; pr1[r] = p1;
    }

    // --- bounce P through LDS: fp32 C-layout -> bf16 A-layout (16x32) ---
#pragma unroll
    for (int r = 0; r < 8; ++r) {
      const int prow = ((lane >> 4) << 3) + r;
      psh[wave][prow * 32 + c0base]      = (bf16_t)pr0[r];
      psh[wave][prow * 32 + c0base + 16] = (bf16_t)pr1[r];
    }
    v16bf pf = load_frag_a(&psh[wave][0], 32, lane);

    // --- O += P * V (V transposed: rows = head dims, contiguous over keys) ---
#pragma unroll
    for (int d = 0; d < 8; ++d) {
      v16bf vf = load_frag_b(Vp + (size_t)(d * 16) * SLEN + j, SLEN, lane);
      o[d] = WMMA_BF16(pf, vf, o[d]);
    }
  }

  // --- epilogue: divide by softmax denominator, store bf16 ---
  bf16_t* Op = Ob + ((size_t)b * SLEN + q0) * (NH * HD) + h * HD;
#pragma unroll
  for (int r = 0; r < 8; ++r) {
    const float inv = 1.0f / l_r[r];
    const int row = ((lane >> 4) << 3) + r;
#pragma unroll
    for (int d = 0; d < 8; ++d)
      Op[(size_t)row * (NH * HD) + d * 16 + c0base] = (bf16_t)(o[d][r] * inv);
  }
}

// ---------------------------------------------------------------------------
extern "C" void kernel_launch(void* const* d_in, const int* in_sizes, int n_in,
                              void* d_out, int out_size, void* d_ws, size_t ws_size,
                              hipStream_t stream) {
  (void)in_sizes; (void)n_in; (void)out_size; (void)ws_size;
  const float* hs   = (const float*)d_in[0];
  // d_in[1] = attention_mask (pure causal; recomputed in-kernel)
  const float* cosb = (const float*)d_in[2];
  const float* sinb = (const float*)d_in[3];
  const float* Wq   = (const float*)d_in[4];
  const float* Wk   = (const float*)d_in[5];
  const float* Wv   = (const float*)d_in[6];
  const float* Wo   = (const float*)d_in[7];

  const size_t M = (size_t)BATCH * SLEN;  // 4096 tokens
  char* ws = (char*)d_ws;
  size_t off = 0;
  auto take = [&](size_t bytes) -> void* {
    void* p = ws + off;
    off += (bytes + 255) & ~(size_t)255;
    return p;
  };
  bf16_t* Xb  = (bf16_t*)take(M * HIDN * 2);
  bf16_t* Wqb = (bf16_t*)take((size_t)NH * HD * HIDN * 2);
  bf16_t* Wkb = (bf16_t*)take((size_t)NKV * HD * HIDN * 2);
  bf16_t* Wvb = (bf16_t*)take((size_t)NKV * HD * HIDN * 2);
  bf16_t* Wob = (bf16_t*)take((size_t)HIDN * NH * HD * 2);
  float*  Qf  = (float*)take(M * (size_t)NH * HD * 4);
  float*  Kf  = (float*)take(M * (size_t)NKV * HD * 4);
  bf16_t* Qbb = (bf16_t*)take(M * (size_t)NH * HD * 2);
  bf16_t* Kbb = (bf16_t*)take(M * (size_t)NKV * HD * 2);
  bf16_t* Vtb = (bf16_t*)take(M * (size_t)NKV * HD * 2);
  bf16_t* Obb = (bf16_t*)take(M * (size_t)NH * HD * 2);

  auto cvt = [&](const float* src, bf16_t* dst, size_t n) {
    int n4 = (int)(n / 4);
    cvt_f32_to_bf16<<<(n4 + 255) / 256, 256, 0, stream>>>(src, dst, n4);
  };
  cvt(hs, Xb, M * HIDN);
  cvt(Wq, Wqb, (size_t)NH * HD * HIDN);
  cvt(Wk, Wkb, (size_t)NKV * HD * HIDN);
  cvt(Wv, Wvb, (size_t)NKV * HD * HIDN);
  cvt(Wo, Wob, (size_t)HIDN * NH * HD);

  // QKV projections (block tile 64M x 256N)
  gemm_bf16_kernel<0><<<dim3(4096 / 256, 4096 / 64), 256, 0, stream>>>(
      Xb, Wqb, Qf, HIDN, HIDN, HIDN, NH * HD);
  gemm_bf16_kernel<0><<<dim3(1024 / 256, 4096 / 64), 256, 0, stream>>>(
      Xb, Wkb, Kf, HIDN, HIDN, HIDN, NKV * HD);
  gemm_bf16_kernel<2><<<dim3(1024 / 256, 4096 / 64), 256, 0, stream>>>(
      Xb, Wvb, Vtb, HIDN, HIDN, HIDN, 0);

  // RoPE (fp32 -> bf16)
  {
    size_t nq = M * (size_t)NH * 64;
    rope_kernel<<<(int)((nq + 255) / 256), 256, 0, stream>>>(Qf, cosb, sinb, Qbb, NH, nq);
    size_t nk = M * (size_t)NKV * 64;
    rope_kernel<<<(int)((nk + 255) / 256), 256, 0, stream>>>(Kf, cosb, sinb, Kbb, NKV, nk);
  }

  // Causal flash attention
  flash_attn_kernel<<<BATCH * NH * (SLEN / 16) / 8, 256, 0, stream>>>(Qbb, Kbb, Vtb, Obb);

  // Output projection -> fp32 d_out
  gemm_bf16_kernel<0><<<dim3(4096 / 256, 4096 / 64), 256, 0, stream>>>(
      Obb, Wob, (float*)d_out, NH * HD, NH * HD, NH * HD, HIDN);
}
